// Classifier_42872363549009
// MI455X (gfx1250) — matrix operand order repaired and checked by reading, not compiled, not run on hardware
//
#include <hip/hip_runtime.h>
#include <hip/hip_bf16.h>

// ---------------------------------------------------------------------------
// MI455X (gfx1250, wave32) implementation.
// All large GEMMs (LSTM steps, convs-as-GEMM, feat@train_hids) run on
// v_wmma_f32_16x16x32_bf16 with f32 accumulation. bf16 B-matrix tiles are
// moved by the Tensor Data Mover (tensor_load_to_lds + s_wait_tensorcnt);
// the 240MB f32 train_hids stream is converted f32->bf16 during LDS staging
// so it is read exactly once (~10us @ 23.3 TB/s lower bound).
// ---------------------------------------------------------------------------

typedef __attribute__((ext_vector_type(16))) __bf16 v16bf;
typedef __attribute__((ext_vector_type(8)))  float  v8f;
typedef __attribute__((ext_vector_type(4)))  unsigned int u32x4;
typedef __attribute__((ext_vector_type(4)))  unsigned int v4u;
typedef __attribute__((ext_vector_type(8)))  int v8i;
typedef __attribute__((ext_vector_type(4)))  int v4i;

union U16B { u32x4 q; __bf16 h[8]; };
union U32B { u32x4 q[2]; __bf16 h[16]; };

constexpr int B_SZ  = 128;   // batch
constexpr int SEQ   = 64;    // sequence length
constexpr int EMB   = 300;   // embedding dim
constexpr int HID   = 300;   // hidden dim
constexpr int EMBP  = 320;   // padded (mult of 32)
constexpr int HIDP  = 320;
constexpr int KCAT  = 640;   // EMBP + HIDP
constexpr int NGATE = 1200;  // 4*HID
constexpr int NFILT = 100;
constexpr int NTRAIN= 50000;
constexpr int NCLS  = 16;
constexpr int NFEAT = 1200;  // EMB + 3*NFILT + 2*HID

__device__ __forceinline__ float sigf(float x)  { return 1.f / (1.f + __expf(-x)); }
__device__ __forceinline__ float tanhfa(float x){ return 2.f / (1.f + __expf(-2.f*x)) - 1.f; }

// ---------------------------------------------------------------------------
// Tensor Data Mover: 2D tile of 16-bit elements, global -> LDS (linear tile).
// D# per CDNA5 ISA ch.8: group0 = {count|flags, lds_addr, global_addr, type=2},
// group1 = {data_size, tensor dims / tile dims / stride}.  OOB reads beyond
// tensor_dim return zero, which covers K/N tail tiles for free.
// Issued by one wave (TDM ignores EXEC); completion via s_wait_tensorcnt.
// ---------------------------------------------------------------------------
__device__ __forceinline__ void tdm_load_tile_b16(const void* gptr, unsigned lds_off,
                                                  unsigned tile_w, unsigned tile_h,
                                                  unsigned row_stride,
                                                  unsigned tensor_w, unsigned tensor_h) {
  unsigned long long ga = (unsigned long long)gptr;
  v4u g0;
  g0[0] = 1u;                                              // count=1, user mode
  g0[1] = lds_off;                                         // lds_addr (bytes)
  g0[2] = (unsigned)ga;                                    // global_addr[31:0]
  g0[3] = (unsigned)((ga >> 32) & 0x01FFFFFFu) | (2u << 30); // addr[56:32] | type=2
  v8i g1;
  g1[0] = (int)(1u << 16);                                 // data_size=1 (2 bytes)
  g1[1] = (int)((tensor_w & 0xFFFFu) << 16);               // tensor_dim0[15:0]
  g1[2] = (int)(((tensor_w >> 16) & 0xFFFFu) |
                ((tensor_h & 0xFFFFu) << 16));             // dim0 hi | dim1 lo
  g1[3] = (int)(((tensor_h >> 16) & 0xFFFFu) |
                ((tile_w & 0xFFFFu) << 16));               // dim1 hi | tile_dim0
  g1[4] = (int)(tile_h & 0xFFFFu);                         // tile_dim1 (tile_dim2=0)
  g1[5] = (int)row_stride;                                 // tensor_dim0_stride lo32
  g1[6] = 0;
  g1[7] = 0;
  v4i z4 = {0, 0, 0, 0};
#if defined(__clang_major__) && (__clang_major__ >= 23)
  v8i z8 = {0, 0, 0, 0, 0, 0, 0, 0};
  __builtin_amdgcn_tensor_load_to_lds(g0, g1, z4, z4, z8, 0);
#else
  __builtin_amdgcn_tensor_load_to_lds(g0, g1, z4, z4, 0);
#endif
}

// low 32 bits of a generic pointer to LDS == LDS byte offset (ISA 10.2)
__device__ __forceinline__ unsigned lds_addr_of(const void* p) {
  return (unsigned)(unsigned long long)p;
}

// ---------------------------------------------------------------------------
// small helper kernels
// ---------------------------------------------------------------------------
__global__ void lens_kernel(const int* __restrict__ x, int* __restrict__ lens) {
  int b = threadIdx.x;
  if (b < B_SZ) {
    int c = 0;
    for (int s = 0; s < SEQ; ++s) c += (x[b * SEQ + s] != 1);
    lens[b] = c;
  }
}

__global__ void fill_zero_kernel(unsigned int* __restrict__ p, long n) {
  long i = (long)blockIdx.x * blockDim.x + threadIdx.x;
  if (i < n) p[i] = 0u;
}

// xs layout: (dir, s, b, EMBP) bf16; dir1 holds the time-reversed sequence.
__global__ void embed_kernel(const int* __restrict__ x, const float* __restrict__ emb,
                             const int* __restrict__ lens, __bf16* __restrict__ xs) {
  long idx = (long)blockIdx.x * blockDim.x + threadIdx.x;
  const long total = 2L * SEQ * B_SZ * EMBP;
  if (idx >= total) return;
  int e = (int)(idx % EMBP);
  long r = idx / EMBP;
  int b = (int)(r % B_SZ); r /= B_SZ;
  int s = (int)(r % SEQ);
  int d = (int)(r / SEQ);
  int pos = s;
  if (d == 1) { pos = lens[b] - 1 - s; if (pos < 0) pos = 0; }
  int tok = x[b * SEQ + pos];
  float v = (e < EMB) ? emb[(long)tok * EMB + e] : 0.f;
  xs[idx] = (__bf16)v;
}

__global__ void mean_kernel(const int* __restrict__ x, const float* __restrict__ emb,
                            float* __restrict__ meanv) {
  int idx = blockIdx.x * blockDim.x + threadIdx.x;
  if (idx >= B_SZ * EMB) return;
  int b = idx / EMB, e = idx % EMB;
  float s = 0.f;
  for (int t = 0; t < SEQ; ++t) s += emb[(long)x[b * SEQ + t] * EMB + e];
  meanv[idx] = s * (1.f / SEQ);
}

// Wcat: (dir, NGATE, KCAT) bf16 = [Wih | 0pad | Whh | 0pad] per gate row.
__global__ void wcat_kernel(const float* __restrict__ Wih_f, const float* __restrict__ Whh_f,
                            const float* __restrict__ Wih_b, const float* __restrict__ Whh_b,
                            __bf16* __restrict__ Wcat) {
  long idx = (long)blockIdx.x * blockDim.x + threadIdx.x;
  if (idx >= 2L * NGATE * KCAT) return;
  int k = (int)(idx % KCAT);
  long r = idx / KCAT;
  int n = (int)(r % NGATE);
  int d = (int)(r / NGATE);
  const float* Wih = d ? Wih_b : Wih_f;
  const float* Whh = d ? Whh_b : Whh_f;
  float v = 0.f;
  if (k < EMB) v = Wih[(long)n * EMB + k];
  else if (k >= EMBP && k < EMBP + HID) v = Whh[(long)n * HID + (k - EMBP)];
  Wcat[idx] = (__bf16)v;
}

// conv weight repack: (FS, D, k) f32 -> (FS, k*600) bf16, K index = dt*600 + c
__global__ void convw_kernel(const float* __restrict__ w, __bf16* __restrict__ dst, int ksz) {
  int idx = blockIdx.x * blockDim.x + threadIdx.x;
  int total = NFILT * 600 * ksz;
  if (idx >= total) return;
  int f = idx / (600 * ksz);
  int rem = idx % (600 * ksz);
  int dt = rem / 600;
  int c  = rem % 600;
  dst[idx] = (__bf16)w[((long)f * 600 + c) * ksz + dt];
}

// ---------------------------------------------------------------------------
// Generic bf16 WMMA GEMM: C[M,N](f32) = A[M,K](bf16) * B[K,N] (+bias)
// B stored N-major (B[n*ldb+k]).  bf16 B tiles are fetched by the TDM into a
// linear [64][32] LDS tile; f32 B (train_hids) is converted during staging.
// A rows may be segmented: row m -> A + (m/rows_per_seg)*seg_stride
//                                     + (m%rows_per_seg)*lda   (conv windows)
// Block: 128 threads (4 wave32), 64x64 tile, K step 32.
// A tile is staged in WMMA fragment order; B fragments read from the linear
// tile are still a single contiguous 32B LDS access per operand half.
// ---------------------------------------------------------------------------
__global__ __launch_bounds__(128)
void gemm_bf16_kernel(const __bf16* __restrict__ A, int lda, int rows_per_seg, long seg_stride,
                      const __bf16* __restrict__ Bbf, const float* __restrict__ Bf32, int ldb,
                      float* __restrict__ C, int ldc, int M, int N, int K,
                      const float* __restrict__ bias) {
  __shared__ __align__(32) __bf16 AsF[2048];
  __shared__ __align__(32) __bf16 Blin[2048];   // linear [64 rows][32 k]
  const int tid = threadIdx.x, lane = tid & 31, wave = tid >> 5;
  const int rm0 = blockIdx.x * 64;
  const int cn0 = blockIdx.y * 64;

  v8f acc[4];
  for (int j = 0; j < 4; ++j)
    for (int v = 0; v < 8; ++v) acc[j][v] = 0.f;

  for (int k0 = 0; k0 < K; k0 += 32) {
    // ---- stage A tile (64 rows x 32 cols) in fragment order ----
    for (int ch = tid; ch < 256; ch += 128) {
      int r  = ch >> 2;
      int kc = (ch & 3) << 3;
      int m  = rm0 + r;
      int k  = k0 + kc;
      U16B u;
      u.q = (u32x4){0u, 0u, 0u, 0u};
      if (m < M) {
        int seg = m / rows_per_seg;
        int rin = m - seg * rows_per_seg;
        const __bf16* src = A + (long)seg * seg_stride + (long)rin * lda + k;
        if (k + 7 < K) {
          u.q = *(const u32x4*)src;
        } else {
          for (int e = 0; e < 8; ++e) u.h[e] = (k + e < K) ? src[e] : (__bf16)0.f;
        }
      }
      int half = kc >> 4;
      int run  = (kc >> 3) & 1;
      *(u32x4*)(AsF + ((r >> 4) * 512 + (half * 16 + (r & 15)) * 16 + run * 8)) = u.q;
    }
    // ---- stage B tile: TDM for bf16 source, VALU convert for f32 source ----
    if (Bf32) {
      int ch   = tid;            // 128 chunks of 16 bf16
      int nrel = ch >> 1;
      int half = ch & 1;
      int n    = cn0 + nrel;
      int kb   = k0 + half * 16;
      U32B u;
      for (int e = 0; e < 16; ++e) {
        int k = kb + e;
        float f = (n < N && k < K) ? Bf32[(long)n * ldb + k] : 0.f;
        u.h[e] = (__bf16)f;
      }
      if (n < N && kb + 32 < K)
        __builtin_prefetch((const void*)(Bf32 + (long)n * ldb + kb + 32), 0, 1);
      u32x4* dst = (u32x4*)(Blin + nrel * 32 + half * 16);
      dst[0] = u.q[0]; dst[1] = u.q[1];
    } else if (tid < 32) {       // one wave drives the TDM (EXEC is ignored)
      tdm_load_tile_b16(Bbf + (long)cn0 * ldb + k0, lds_addr_of(Blin),
                        32u, 64u, (unsigned)ldb,
                        (unsigned)(K - k0),
                        (unsigned)(N > cn0 ? N - cn0 : 0));
      __builtin_amdgcn_s_wait_tensorcnt(0);
    }
    __syncthreads();
    int bq = (lane & 15) * 32 + (lane >> 4) * 16;
    v16bf a  = *(const v16bf*)(AsF + wave * 512 + lane * 16);
    v16bf b0 = *(const v16bf*)(Blin +    0 + bq);
    v16bf b1 = *(const v16bf*)(Blin +  512 + bq);
    v16bf b2 = *(const v16bf*)(Blin + 1024 + bq);
    v16bf b3 = *(const v16bf*)(Blin + 1536 + bq);
    acc[0] = __builtin_amdgcn_wmma_f32_16x16x32_bf16(false, a, false, b0, (short)0, acc[0], false, false);
    acc[1] = __builtin_amdgcn_wmma_f32_16x16x32_bf16(false, a, false, b1, (short)0, acc[1], false, false);
    acc[2] = __builtin_amdgcn_wmma_f32_16x16x32_bf16(false, a, false, b2, (short)0, acc[2], false, false);
    acc[3] = __builtin_amdgcn_wmma_f32_16x16x32_bf16(false, a, false, b3, (short)0, acc[3], false, false);
    __syncthreads();
  }
  // epilogue: VGPR v of a 16x16 C tile holds row v (lanes 0-15) / 8+v (16-31)
  int mrow  = rm0 + wave * 16 + ((lane >> 4) << 3);
  int ncol0 = cn0 + (lane & 15);
  for (int j = 0; j < 4; ++j) {
    int n = ncol0 + j * 16;
    if (n < N) {
      float bv = bias ? bias[n] : 0.f;
      for (int v = 0; v < 8; ++v) {
        int m = mrow + v;
        if (m < M) C[(long)m * ldc + n] = acc[j][v] + bv;
      }
    }
  }
}

// ---------------------------------------------------------------------------
// LSTM step GEMM: gates[d,b,n] = [x_t | h] @ [Wih | Whh]^T + bias
// M=128, N=1200, K=640.  grid = (2 Mtiles, 19 Ntiles, 2 dirs)
// Weight tiles stream through the TDM; activations staged in fragment order.
// ---------------------------------------------------------------------------
__global__ __launch_bounds__(128)
void lstm_gemm_kernel(const __bf16* __restrict__ xs, const __bf16* __restrict__ hbf,
                      const __bf16* __restrict__ Wcat,
                      const float* __restrict__ bias_f, const float* __restrict__ bias_b,
                      float* __restrict__ gates, int t) {
  __shared__ __align__(32) __bf16 AsF[2048];
  __shared__ __align__(32) __bf16 Blin[2048];
  const int tid = threadIdx.x, lane = tid & 31, wave = tid >> 5;
  const int rm0 = blockIdx.x * 64;
  const int cn0 = blockIdx.y * 64;
  const int d   = blockIdx.z;
  const __bf16* Wd  = Wcat + (long)d * NGATE * KCAT;
  const __bf16* xsd = xs + ((long)(d * SEQ + t) * B_SZ) * EMBP;
  const __bf16* hd  = hbf + (long)d * B_SZ * HIDP;
  const float*  bias = d ? bias_b : bias_f;
  float* Cd = gates + (long)d * B_SZ * NGATE;

  v8f acc[4];
  for (int j = 0; j < 4; ++j)
    for (int v = 0; v < 8; ++v) acc[j][v] = 0.f;

  for (int k0 = 0; k0 < KCAT; k0 += 32) {
    for (int ch = tid; ch < 256; ch += 128) {
      int r  = ch >> 2;
      int kc = (ch & 3) << 3;
      int m  = rm0 + r;                 // always < 128
      int k  = k0 + kc;                 // chunks never straddle the 320 split
      const __bf16* src = (k < EMBP) ? (xsd + (long)m * EMBP + k)
                                     : (hd  + (long)m * HIDP + (k - EMBP));
      U16B u;
      u.q = *(const u32x4*)src;
      int half = kc >> 4;
      int run  = (kc >> 3) & 1;
      *(u32x4*)(AsF + ((r >> 4) * 512 + (half * 16 + (r & 15)) * 16 + run * 8)) = u.q;
    }
    if (tid < 32) {                     // wave 0 drives the TDM weight fetch
      tdm_load_tile_b16(Wd + (long)cn0 * KCAT + k0, lds_addr_of(Blin),
                        32u, 64u, (unsigned)KCAT,
                        (unsigned)(KCAT - k0),
                        (unsigned)(NGATE > cn0 ? NGATE - cn0 : 0));
      __builtin_amdgcn_s_wait_tensorcnt(0);
    }
    __syncthreads();
    int bq = (lane & 15) * 32 + (lane >> 4) * 16;
    v16bf a  = *(const v16bf*)(AsF + wave * 512 + lane * 16);
    v16bf b0 = *(const v16bf*)(Blin +    0 + bq);
    v16bf b1 = *(const v16bf*)(Blin +  512 + bq);
    v16bf b2 = *(const v16bf*)(Blin + 1024 + bq);
    v16bf b3 = *(const v16bf*)(Blin + 1536 + bq);
    acc[0] = __builtin_amdgcn_wmma_f32_16x16x32_bf16(false, a, false, b0, (short)0, acc[0], false, false);
    acc[1] = __builtin_amdgcn_wmma_f32_16x16x32_bf16(false, a, false, b1, (short)0, acc[1], false, false);
    acc[2] = __builtin_amdgcn_wmma_f32_16x16x32_bf16(false, a, false, b2, (short)0, acc[2], false, false);
    acc[3] = __builtin_amdgcn_wmma_f32_16x16x32_bf16(false, a, false, b3, (short)0, acc[3], false, false);
    __syncthreads();
  }
  int mrow  = rm0 + wave * 16 + ((lane >> 4) << 3);
  int ncol0 = cn0 + (lane & 15);
  for (int j = 0; j < 4; ++j) {
    int n = ncol0 + j * 16;
    if (n < NGATE) {
      float bv = bias[n];
      for (int v = 0; v < 8; ++v)
        Cd[(long)(mrow + v) * NGATE + n] = acc[j][v] + bv;
    }
  }
}

// gate nonlinearities + state update + scatter into conv input
__global__ void lstm_point_kernel(const float* __restrict__ gates,
                                  float* __restrict__ h, float* __restrict__ c,
                                  __bf16* __restrict__ hbf, __bf16* __restrict__ convin,
                                  const int* __restrict__ lens, int t) {
  int idx = blockIdx.x * blockDim.x + threadIdx.x;
  if (idx >= 2 * B_SZ * HID) return;
  int d = idx / (B_SZ * HID);
  int r = idx % (B_SZ * HID);
  int b = r / HID;
  int j = r % HID;
  const float* g = gates + (long)(d * B_SZ + b) * NGATE;
  float si = sigf(g[j]);
  float sf = sigf(g[HID + j]);
  float tg = tanhfa(g[2 * HID + j]);
  float so = sigf(g[3 * HID + j]);
  long hc = (long)(d * B_SZ + b) * HID + j;
  float cold = c[hc], hold = h[hc];
  float cn = sf * cold + si * tg;
  float hn = so * tanhfa(cn);
  int L = lens[b];
  bool m = (t < L);
  float h2 = m ? hn : hold;
  float c2 = m ? cn : cold;
  h[hc] = h2;
  c[hc] = c2;
  hbf[(long)(d * B_SZ + b) * HIDP + j] = (__bf16)h2;
  if (m) {
    int pos = (d == 0) ? t : (L - 1 - t);   // un-reverse backward outputs
    convin[((long)b * SEQ + pos) * 600 + d * HID + j] = (__bf16)hn;
  }
}

// max over time + bias + relu  ->  fv[:, off:off+100]
__global__ void conv_reduce_kernel(const float* __restrict__ y, const float* __restrict__ bias,
                                   float* __restrict__ fv, int Tout, int off) {
  int idx = blockIdx.x * blockDim.x + threadIdx.x;
  if (idx >= B_SZ * NFILT) return;
  int b = idx / NFILT, f = idx % NFILT;
  float m = -1e30f;
  for (int t = 0; t < Tout; ++t)
    m = fmaxf(m, y[(long)(b * Tout + t) * NFILT + f]);
  m += bias[f];
  fv[b * 300 + off + f] = fmaxf(m, 0.f);
}

// feat = [mean_emb | relu(fv) | h_f | h_b]  (f32 + bf16 copies)
__global__ void feat_kernel(const float* __restrict__ meanv, const float* __restrict__ fv,
                            const float* __restrict__ h,
                            float* __restrict__ featf, __bf16* __restrict__ featbf) {
  int idx = blockIdx.x * blockDim.x + threadIdx.x;
  if (idx >= B_SZ * NFEAT) return;
  int b = idx / NFEAT, k = idx % NFEAT;
  float v;
  if      (k < 300) v = meanv[b * 300 + k];
  else if (k < 600) v = fv[b * 300 + (k - 300)];
  else if (k < 900) v = h[(long)(0 * B_SZ + b) * HID + (k - 600)];
  else              v = h[(long)(1 * B_SZ + b) * HID + (k - 900)];
  featf[idx]  = v;
  featbf[idx] = (__bf16)v;
}

// per-row softmax over 50000 + pred = weights@train_ans + final linear blend
__global__ __launch_bounds__(256)
void softmax_pred_kernel(const float* __restrict__ scores, const float* __restrict__ ta,
                         const float* __restrict__ featf, const float* __restrict__ Wout,
                         const float* __restrict__ bout, float* __restrict__ out) {
  int b = blockIdx.x;
  int tid = threadIdx.x;
  __shared__ float red[256];
  __shared__ float accs[NCLS];
  const float* srow = scores + (long)b * NTRAIN;
  float m = -1e30f;
  for (int n = tid; n < NTRAIN; n += 256) m = fmaxf(m, srow[n]);
  red[tid] = m; __syncthreads();
  for (int s = 128; s > 0; s >>= 1) { if (tid < s) red[tid] = fmaxf(red[tid], red[tid + s]); __syncthreads(); }
  m = red[0]; __syncthreads();

  float sum = 0.f;
  float acc[NCLS];
  for (int cc = 0; cc < NCLS; ++cc) acc[cc] = 0.f;
  for (int n = tid; n < NTRAIN; n += 256) {
    float e = __expf(srow[n] - m);
    sum += e;
    const float* tr = ta + (long)n * NCLS;
    #pragma unroll
    for (int cc = 0; cc < NCLS; ++cc) acc[cc] += e * tr[cc];
  }
  red[tid] = sum; __syncthreads();
  for (int s = 128; s > 0; s >>= 1) { if (tid < s) red[tid] += red[tid + s]; __syncthreads(); }
  sum = red[0]; __syncthreads();
  for (int cc = 0; cc < NCLS; ++cc) {
    red[tid] = acc[cc]; __syncthreads();
    for (int s = 128; s > 0; s >>= 1) { if (tid < s) red[tid] += red[tid + s]; __syncthreads(); }
    if (tid == 0) accs[cc] = red[0];
    __syncthreads();
  }
  if (tid < NCLS) {
    int cc = tid;
    float lin = bout[cc];
    const float* fr = featf + (long)b * NFEAT;
    const float* wr = Wout  + (long)cc * NFEAT;
    for (int k = 0; k < NFEAT; ++k) lin += fr[k] * wr[k];
    out[b * NCLS + cc] = 0.5f * (accs[cc] / sum) + 0.5f * lin;
  }
}

// ---------------------------------------------------------------------------
extern "C" void kernel_launch(void* const* d_in, const int* in_sizes, int n_in,
                              void* d_out, int out_size, void* d_ws, size_t ws_size,
                              hipStream_t stream) {
  const int*   x     = (const int*)  d_in[0];
  const float* embed = (const float*)d_in[1];
  const float* Wih_f = (const float*)d_in[2];
  const float* Whh_f = (const float*)d_in[3];
  const float* b_f   = (const float*)d_in[4];
  const float* Wih_b = (const float*)d_in[5];
  const float* Whh_b = (const float*)d_in[6];
  const float* b_b   = (const float*)d_in[7];
  const float* cw3   = (const float*)d_in[8];
  const float* cb3   = (const float*)d_in[9];
  const float* cw4   = (const float*)d_in[10];
  const float* cb4   = (const float*)d_in[11];
  const float* cw5   = (const float*)d_in[12];
  const float* cb5   = (const float*)d_in[13];
  const float* Wout  = (const float*)d_in[14];
  const float* bout  = (const float*)d_in[15];
  const float* thids = (const float*)d_in[16];
  const float* tans  = (const float*)d_in[17];
  float* out = (float*)d_out;
  (void)in_sizes; (void)n_in; (void)out_size; (void)ws_size;

  char* ws = (char*)d_ws;
  size_t off = 0;
  auto take = [&](size_t bytes) {
    char* p = ws + off;
    off = (off + bytes + 255) & ~((size_t)255);
    return p;
  };
  __bf16* xs     = (__bf16*)take(2ull * SEQ * B_SZ * EMBP * 2);     // ~10.5 MB
  __bf16* Wcat   = (__bf16*)take(2ull * NGATE * KCAT * 2);          // ~3 MB
  __bf16* cwb3   = (__bf16*)take((size_t)NFILT * 1800 * 2);
  __bf16* cwb4   = (__bf16*)take((size_t)NFILT * 2400 * 2);
  __bf16* cwb5   = (__bf16*)take((size_t)NFILT * 3000 * 2);
  __bf16* convin = (__bf16*)take((size_t)B_SZ * SEQ * 600 * 2);     // ~9.8 MB
  float*  gates  = (float*) take(2ull * B_SZ * NGATE * 4);
  float*  hbuf   = (float*) take(2ull * B_SZ * HID * 4);
  float*  cbuf   = (float*) take(2ull * B_SZ * HID * 4);
  __bf16* hbf    = (__bf16*)take(2ull * B_SZ * HIDP * 2);
  float*  ybuf   = (float*) take((size_t)B_SZ * 62 * NFILT * 4);
  float*  fv     = (float*) take((size_t)B_SZ * 300 * 4);
  float*  meanv  = (float*) take((size_t)B_SZ * 300 * 4);
  int*    lens   = (int*)   take((size_t)B_SZ * 4);
  float*  featf  = (float*) take((size_t)B_SZ * NFEAT * 4);
  __bf16* featbf = (__bf16*)take((size_t)B_SZ * NFEAT * 2);
  float*  scores = (float*) take((size_t)B_SZ * NTRAIN * 4);        // ~25.6 MB

  lens_kernel<<<1, 128, 0, stream>>>(x, lens);

  {   // zero conv input, h, c, hbf (incl. bf16 pad columns)
    long n1 = (long)B_SZ * SEQ * 600 * 2 / 4;
    fill_zero_kernel<<<(unsigned)((n1 + 255) / 256), 256, 0, stream>>>((unsigned int*)convin, n1);
    long n2 = 2L * B_SZ * HID;
    fill_zero_kernel<<<(unsigned)((n2 + 255) / 256), 256, 0, stream>>>((unsigned int*)hbuf, n2);
    fill_zero_kernel<<<(unsigned)((n2 + 255) / 256), 256, 0, stream>>>((unsigned int*)cbuf, n2);
    long n3 = 2L * B_SZ * HIDP * 2 / 4;
    fill_zero_kernel<<<(unsigned)((n3 + 255) / 256), 256, 0, stream>>>((unsigned int*)hbf, n3);
  }

  {
    long te = 2L * SEQ * B_SZ * EMBP;
    embed_kernel<<<(unsigned)((te + 255) / 256), 256, 0, stream>>>(x, embed, lens, xs);
  }
  mean_kernel<<<(B_SZ * EMB + 255) / 256, 256, 0, stream>>>(x, embed, meanv);
  {
    long tw = 2L * NGATE * KCAT;
    wcat_kernel<<<(unsigned)((tw + 255) / 256), 256, 0, stream>>>(Wih_f, Whh_f, Wih_b, Whh_b, Wcat);
  }
  convw_kernel<<<(NFILT * 600 * 3 + 255) / 256, 256, 0, stream>>>(cw3, cwb3, 3);
  convw_kernel<<<(NFILT * 600 * 4 + 255) / 256, 256, 0, stream>>>(cw4, cwb4, 4);
  convw_kernel<<<(NFILT * 600 * 5 + 255) / 256, 256, 0, stream>>>(cw5, cwb5, 5);

  // ---- BiLSTM: 64 sequential (graph-captured) step pairs ----
  for (int t = 0; t < SEQ; ++t) {
    lstm_gemm_kernel<<<dim3(2, 19, 2), 128, 0, stream>>>(xs, hbf, Wcat, b_f, b_b, gates, t);
    lstm_point_kernel<<<(2 * B_SZ * HID + 255) / 256, 256, 0, stream>>>(gates, hbuf, cbuf, hbf, convin, lens, t);
  }

  // ---- conv1d (k=3,4,5) as implicit GEMMs + max/relu reductions ----
  gemm_bf16_kernel<<<dim3(124, 2), 128, 0, stream>>>(convin, 600, 62, (long)SEQ * 600,
      cwb3, nullptr, 1800, ybuf, NFILT, B_SZ * 62, NFILT, 1800, nullptr);
  conv_reduce_kernel<<<(B_SZ * NFILT + 255) / 256, 256, 0, stream>>>(ybuf, cb3, fv, 62, 0);
  gemm_bf16_kernel<<<dim3(122, 2), 128, 0, stream>>>(convin, 600, 61, (long)SEQ * 600,
      cwb4, nullptr, 2400, ybuf, NFILT, B_SZ * 61, NFILT, 2400, nullptr);
  conv_reduce_kernel<<<(B_SZ * NFILT + 255) / 256, 256, 0, stream>>>(ybuf, cb4, fv, 61, 100);
  gemm_bf16_kernel<<<dim3(120, 2), 128, 0, stream>>>(convin, 600, 60, (long)SEQ * 600,
      cwb5, nullptr, 3000, ybuf, NFILT, B_SZ * 60, NFILT, 3000, nullptr);
  conv_reduce_kernel<<<(B_SZ * NFILT + 255) / 256, 256, 0, stream>>>(ybuf, cb5, fv, 60, 200);

  feat_kernel<<<(B_SZ * NFEAT + 255) / 256, 256, 0, stream>>>(meanv, fv, hbuf, featf, featbf);

  // ---- scores = feat @ train_hids.T : the 240MB memory-bound GEMM ----
  gemm_bf16_kernel<<<dim3(2, 782), 128, 0, stream>>>(featbf, NFEAT, B_SZ, 0L,
      nullptr, thids, NFEAT, scores, NTRAIN, B_SZ, NTRAIN, NFEAT, nullptr);

  softmax_pred_kernel<<<B_SZ, 256, 0, stream>>>(scores, tans, featf, Wout, bout, out);
}
